// Attention_87110526697973
// MI455X (gfx1250) — compile-verified, matrix operand order checked by reading, not run on hardware
//
#include <hip/hip_runtime.h>
#include <cmath>

// ---------------------------------------------------------------------------
// MI455X / gfx1250 fused DiT attention block.
// bf16 WMMA (v_wmma_f32_16x16x32_bf16) for all three matmul stages,
// async global->LDS DMA (ASYNCcnt) with double-buffered tiles,
// f32 statistics for LayerNorm / RoPE / softmax.
// ---------------------------------------------------------------------------

typedef __bf16 bf16_t;
typedef __attribute__((ext_vector_type(2)))  bf16_t v2bf;
typedef __attribute__((ext_vector_type(8)))  bf16_t v8bf;
typedef __attribute__((ext_vector_type(16))) bf16_t v16bf;
typedef __attribute__((ext_vector_type(8)))  float  v8f;

#define Bc 8
#define Tc 1024
#define Dc 1024
#define Hc 16
#define Ec 64
#define N3c 3072   // 3*D

// ---- CDNA5 async / wait plumbing ------------------------------------------
#if defined(__has_builtin)
#if __has_builtin(__builtin_amdgcn_global_load_async_to_lds_b128)
#define HAVE_ASYNC_B 1
#endif
#if __has_builtin(__builtin_amdgcn_s_wait_asynccnt)
#define HAVE_WAIT_ASYNC_B 1
#endif
#if __has_builtin(__builtin_amdgcn_s_wait_dscnt)
#define HAVE_WAIT_DS_B 1
#endif
#endif
#ifndef HAVE_ASYNC_B
#define HAVE_ASYNC_B 0
#endif
#ifndef HAVE_WAIT_ASYNC_B
#define HAVE_WAIT_ASYNC_B 0
#endif
#ifndef HAVE_WAIT_DS_B
#define HAVE_WAIT_DS_B 0
#endif

// b128 payload type of the async builtin: pointer-to-v4i per the clang
// signature (AS1 source, AS3 destination).
typedef int v4i_t __attribute__((vector_size(16)));
typedef __attribute__((address_space(1))) v4i_t g_v4i;
typedef __attribute__((address_space(3))) v4i_t l_v4i;

// 16-byte async global->LDS copy (GLOBAL_LOAD_ASYNC_TO_LDS_B128, ASYNCcnt).
// Generic shared-pointer low 32 bits == wave-relative LDS address (ISA 10.2).
__device__ __forceinline__ void async_copy16(void* lds_dst, const void* gsrc) {
#if HAVE_ASYNC_B
  __builtin_amdgcn_global_load_async_to_lds_b128(
      (g_v4i*)(uintptr_t)gsrc,
      (l_v4i*)(unsigned)(uintptr_t)lds_dst, 0, 0);
#else
  unsigned l = (unsigned)(uintptr_t)lds_dst;
  unsigned long long g = (unsigned long long)(uintptr_t)gsrc;
  asm volatile("global_load_async_to_lds_b128 %0, %1, off"
               :: "v"(l), "v"(g) : "memory");
#endif
}

__device__ __forceinline__ void wait_async0() {
#if HAVE_WAIT_ASYNC_B
  __builtin_amdgcn_s_wait_asynccnt(0);
#else
  asm volatile("s_wait_asynccnt 0x0" ::: "memory");
#endif
}

__device__ __forceinline__ void wait_ds0() {
#if HAVE_WAIT_DS_B
  __builtin_amdgcn_s_wait_dscnt(0);
#else
  asm volatile("s_wait_dscnt 0x0" ::: "memory");
#endif
}

// ---------------------------------------------------------------------------
__device__ __forceinline__ v8f vzero8() {
  v8f z = {0.f, 0.f, 0.f, 0.f, 0.f, 0.f, 0.f, 0.f};
  return z;
}

// Build a 16x32 bf16 A/B fragment for one lane.
// Per ISA 7.12.2: lane<16 holds K = {0..7, 16..23}, lane>=16 holds {8..15, 24..31}.
// Caller passes p = &row[ko] with ko = (lane<16)?0:8 -> two contiguous 16B loads.
__device__ __forceinline__ v16bf load_frag(const bf16_t* p) {
  v8bf lo = *(const v8bf*)(p);
  v8bf hi = *(const v8bf*)(p + 16);
  v16bf r;
#pragma unroll
  for (int i = 0; i < 8; ++i) { r[i] = lo[i]; r[i + 8] = hi[i]; }
  return r;
}

__device__ __forceinline__ v8f wmma_bf16(v16bf a, v16bf b, v8f c) {
  return __builtin_amdgcn_wmma_f32_16x16x32_bf16(false, a, false, b, (short)0, c,
                                                 false, false);
}

// ---------------------------------------------------------------------------
// Small elementwise helpers
// ---------------------------------------------------------------------------
__global__ void silu_kernel(const float* __restrict__ in, float* __restrict__ out,
                            int n) {
  int i = blockIdx.x * 256 + threadIdx.x;
  if (i < n) { float v = in[i]; out[i] = v / (1.f + __expf(-v)); }
}

__global__ void cvt_bf16_kernel(const float* __restrict__ in,
                                bf16_t* __restrict__ out, int n) {
  int i = blockIdx.x * 256 + threadIdx.x;
  if (i < n) out[i] = (bf16_t)in[i];
}

// mod = silu(time) @ mod_w + mod_b   (8 x 1024 x 3072, tiny)
__global__ __launch_bounds__(256)
void mod_kernel(const float* __restrict__ st, const float* __restrict__ mod_w,
                const float* __restrict__ mod_b, float* __restrict__ mod) {
  int n = blockIdx.x * 256 + threadIdx.x;
  int b = blockIdx.y;
  const float* sb = st + (size_t)b * 1024;
  float acc = 0.f;
  for (int k = 0; k < 1024; ++k)
    acc = fmaf(sb[k], mod_w[(size_t)k * N3c + n], acc);
  mod[(size_t)b * N3c + n] = acc + mod_b[n];
}

// ---------------------------------------------------------------------------
// LayerNorm(x) * (scale+1) + shift  ->  bf16 h.  One block per token row.
// ---------------------------------------------------------------------------
__global__ __launch_bounds__(256)
void ln_mod_kernel(const float* __restrict__ x, const float* __restrict__ mod,
                   bf16_t* __restrict__ h) {
  int row = blockIdx.x;              // 0..B*T-1
  int tid = threadIdx.x;
  int b   = row >> 10;               // T = 1024
  const float* xr = x + (size_t)row * Dc;
  float v[4];
  float s = 0.f, ss = 0.f;
#pragma unroll
  for (int i = 0; i < 4; ++i) {
    v[i] = xr[tid + i * 256];
    s += v[i]; ss += v[i] * v[i];
  }
#pragma unroll
  for (int m = 16; m >= 1; m >>= 1) { s += __shfl_xor(s, m); ss += __shfl_xor(ss, m); }
  __shared__ float red[16];
  int lane = tid & 31, w = tid >> 5;
  if (lane == 0) { red[w] = s; red[8 + w] = ss; }
  __syncthreads();
  float ts = 0.f, tss = 0.f;
#pragma unroll
  for (int i = 0; i < 8; ++i) { ts += red[i]; tss += red[8 + i]; }
  float mu   = ts * (1.f / Dc);
  float var  = tss * (1.f / Dc) - mu * mu;
  float rstd = rsqrtf(var + 1e-6f);
  const float* mb = mod + (size_t)b * N3c;
#pragma unroll
  for (int i = 0; i < 4; ++i) {
    int d = tid + i * 256;
    float y = (v[i] - mu) * rstd;
    y = y * (mb[Dc + d] + 1.f) + mb[d];          // scale+1, shift
    h[(size_t)row * Dc + d] = (bf16_t)y;
  }
}

// ---------------------------------------------------------------------------
// bf16 tiled GEMM: C(MxN) = A(MxK) * B(KxN), 128x128x32 tiles, 8 waves,
// double-buffered LDS: A tile via async DMA, B tile via wide transpose stores.
// mode 0: scatter fp32 into qkv buffer (3,B,H,T,e)
// mode 1: out[r][n] = acc * gate[b][n]   (gate = mod[b][2D + n])
// ---------------------------------------------------------------------------
__global__ __launch_bounds__(256)
void gemm_bf16_kernel(const bf16_t* __restrict__ A, const bf16_t* __restrict__ Bm,
                      int M, int N, int K, int mode,
                      float* __restrict__ dst, const float* __restrict__ mod) {
  __shared__ bf16_t As[2][128][32];
  __shared__ bf16_t Bs[2][128][32];     // transposed: Bs[buf][n][k]

  int tid  = threadIdx.x;
  int lane = tid & 31;
  int wave = tid >> 5;               // 0..7
  int wm   = wave >> 2;              // 0..1 : 64 rows each
  int wn   = wave & 3;               // 0..3 : 32 cols each
  int m0   = blockIdx.y * 128;
  int n0   = blockIdx.x * 128;
  int ar   = lane & 15;
  int ko   = (lane & 16) ? 8 : 0;

  auto stage = [&](int buf, int kb) {
    // A tile 128x32: 512 16B chunks, async DMA straight into LDS
#pragma unroll
    for (int c = tid; c < 512; c += 256) {
      int row = c >> 2, kc = (c & 3) * 8;
      async_copy16(&As[buf][row][kc], &A[(size_t)(m0 + row) * K + kb + kc]);
    }
    // B tile 32x128 -> transposed: each thread handles 8 n-cols x 2 k-rows
    int kp = (tid & 15) * 2;
    int nb = (tid >> 4) * 8;
    v8bf r0 = *(const v8bf*)&Bm[(size_t)(kb + kp) * N + n0 + nb];
    v8bf r1 = *(const v8bf*)&Bm[(size_t)(kb + kp + 1) * N + n0 + nb];
#pragma unroll
    for (int i = 0; i < 8; ++i) {
      v2bf p; p[0] = r0[i]; p[1] = r1[i];
      *(v2bf*)&Bs[buf][nb + i][kp] = p;
    }
  };

  v8f acc[4][2];
#pragma unroll
  for (int i = 0; i < 4; ++i)
#pragma unroll
    for (int j = 0; j < 2; ++j) acc[i][j] = vzero8();

  stage(0, 0);
  int cur = 0;
  for (int kb = 0; kb < K; kb += 32) {
    wait_async0();                   // own async copies for tile 'cur' done
    __syncthreads();                 // everyone's copies/stores visible
    if (kb + 32 < K) stage(cur ^ 1, kb + 32);   // overlap DMA with WMMAs

    v16bf af[4], bfr[2];
#pragma unroll
    for (int i = 0; i < 4; ++i)
      af[i] = load_frag(&As[cur][wm * 64 + i * 16 + ar][ko]);
#pragma unroll
    for (int j = 0; j < 2; ++j)
      bfr[j] = load_frag(&Bs[cur][wn * 32 + j * 16 + ar][ko]);
#pragma unroll
    for (int i = 0; i < 4; ++i)
#pragma unroll
      for (int j = 0; j < 2; ++j)
        acc[i][j] = wmma_bf16(af[i], bfr[j], acc[i][j]);
    cur ^= 1;
  }

  // Epilogue (C layout: lane = N col, VGPR r = M row, +8 for lanes 16-31)
  int rofs = (lane & 16) ? 8 : 0;
#pragma unroll
  for (int i = 0; i < 4; ++i) {
    int gmb = m0 + wm * 64 + i * 16 + rofs;
#pragma unroll
    for (int j = 0; j < 2; ++j) {
      int gn = n0 + wn * 32 + j * 16 + ar;
      if (mode == 0) {
        int which = gn >> 10;
        int d = gn & 1023;
        int hh = d >> 6, ei = d & 63;
#pragma unroll
        for (int r = 0; r < 8; ++r) {
          int gm = gmb + r;
          int bb = gm >> 10, tt = gm & 1023;
          size_t off = ((((size_t)which * Bc + bb) * Hc + hh) * Tc + tt) * Ec + ei;
          dst[off] = acc[i][j][r];
        }
      } else {
#pragma unroll
        for (int r = 0; r < 8; ++r) {
          int gm = gmb + r;
          int bb = gm >> 10;
          float g = mod[(size_t)bb * N3c + 2 * Dc + gn];
          dst[(size_t)gm * N + gn] = acc[i][j][r] * g;
        }
      }
    }
  }
}

// ---------------------------------------------------------------------------
// Per-head LayerNorm + RoPE on q,k; convert v.  One wave per (b,h,t) row.
// ---------------------------------------------------------------------------
__global__ __launch_bounds__(256)
void qk_rope_v_kernel(const float* __restrict__ qkv, const float* __restrict__ pos,
                      const float* __restrict__ qw, const float* __restrict__ kw,
                      bf16_t* __restrict__ qhat, bf16_t* __restrict__ khat,
                      bf16_t* __restrict__ vhat) {
  const size_t TOT = (size_t)Bc * Hc * Tc * Ec;
  int gid  = blockIdx.x * 8 + (threadIdx.x >> 5);   // row over (B,H,T)
  int lane = threadIdx.x & 31;
  int t    = gid & (Tc - 1);
  size_t base = (size_t)gid * Ec;

  { // v: straight convert
    const float* vs = qkv + 2 * TOT + base;
    vhat[base + lane]      = (bf16_t)vs[lane];
    vhat[base + 32 + lane] = (bf16_t)vs[32 + lane];
  }
#pragma unroll
  for (int which = 0; which < 2; ++which) {
    const float* src = qkv + (size_t)which * TOT + base;
    const float* wgt = which ? kw : qw;
    bf16_t* dstp = (which ? khat : qhat) + base;
    float x0 = src[lane], x1 = src[32 + lane];
    float s = x0 + x1, ss = x0 * x0 + x1 * x1;
#pragma unroll
    for (int m = 16; m >= 1; m >>= 1) { s += __shfl_xor(s, m); ss += __shfl_xor(ss, m); }
    float mu   = s * (1.f / 64.f);
    float var  = ss * (1.f / 64.f) - mu * mu;
    float rstd = rsqrtf(var + 1e-6f);
    float y0 = (x0 - mu) * rstd * wgt[lane];        // rope half (e < 32)
    float y1 = (x1 - mu) * rstd * wgt[32 + lane];   // pass-through half
    float py = __shfl_xor(y0, 1);                   // pair partner
    int i = lane >> 1;
    float c  = pos[((size_t)t * 16 + i) * 2 + 0];
    float sn = pos[((size_t)t * 16 + i) * 2 + 1];
    float r0 = (lane & 1) ? (y0 * c + py * sn) : (y0 * c - py * sn);
    dstp[lane]      = (bf16_t)r0;
    dstp[32 + lane] = (bf16_t)y1;
  }
}

// ---------------------------------------------------------------------------
// Flash attention. Block = (b,h) x 64 q-rows; 4 waves, each owns 16 q rows.
// K tiles async-DMA'd to LDS, V tiles wide-transposed, double-buffered.
// S = Q K^T via 2 chained WMMAs (K=64), online softmax in C-layout,
// P bounced through LDS to A-layout, PV via 4 WMMAs per 32-key block.
// ---------------------------------------------------------------------------
__global__ __launch_bounds__(128)
void attn_kernel(const bf16_t* __restrict__ qhat, const bf16_t* __restrict__ khat,
                 const bf16_t* __restrict__ vhat, bf16_t* __restrict__ obuf) {
  __shared__ bf16_t Ks[2][32][64];       // key block, row-major
  __shared__ bf16_t Vt[2][64][32];       // value block, transposed: Vt[e][k]
  __shared__ bf16_t Ps[4][16][32];       // per-wave P bounce buffer

  int tid  = threadIdx.x;
  int lane = tid & 31;
  int w    = tid >> 5;                   // wave 0..3
  int qblocks = Tc / 64;
  int bh = blockIdx.x / qblocks;         // 0..B*H-1
  int q0 = (blockIdx.x % qblocks) * 64;
  size_t head_base = (size_t)bh * Tc * Ec;

  int ar = lane & 15;
  int ko = (lane & 16) ? 8 : 0;

  auto stage = [&](int buf, int kb) {
    // K block 32x64 (contiguous in memory): async DMA, 256 16B chunks
#pragma unroll
    for (int c = tid; c < 256; c += 128) {
      int r = c >> 3, cc = (c & 7) * 8;
      async_copy16(&Ks[buf][r][cc],
                   &khat[head_base + (size_t)(kb + r) * Ec + cc]);
    }
    // V block transposed: each thread 8 e-cols x 2 k-rows
    int kp = (tid & 15) * 2;
    int eb = (tid >> 4) * 8;
    v8bf r0 = *(const v8bf*)&vhat[head_base + (size_t)(kb + kp) * Ec + eb];
    v8bf r1 = *(const v8bf*)&vhat[head_base + (size_t)(kb + kp + 1) * Ec + eb];
#pragma unroll
    for (int i = 0; i < 8; ++i) {
      v2bf p; p[0] = r0[i]; p[1] = r1[i];
      *(v2bf*)&Vt[buf][eb + i][kp] = p;
    }
  };

  // Q fragments for this wave's 16 rows (two K=32 fragments over e)
  const bf16_t* qrow = qhat + head_base + (size_t)(q0 + w * 16 + ar) * Ec;
  v16bf qf0 = load_frag(qrow + ko);
  v16bf qf1 = load_frag(qrow + 32 + ko);

  v8f o0 = vzero8(), o1 = vzero8(), o2 = vzero8(), o3 = vzero8();
  float mrow[8], lrow[8];
#pragma unroll
  for (int j = 0; j < 8; ++j) { mrow[j] = -INFINITY; lrow[j] = 0.f; }

  stage(0, 0);
  int cur = 0;
  for (int kb = 0; kb < Tc; kb += 32) {
    wait_async0();
    __syncthreads();
    if (kb + 32 < Tc) stage(cur ^ 1, kb + 32);   // overlap with compute

    // S tiles for two 16-key subtiles (B-frag of Q*K^T == A-pattern over K rows)
    v8f s0 = vzero8(), s1 = vzero8();
    s0 = wmma_bf16(qf0, load_frag(&Ks[cur][ar][ko]),           s0);
    s0 = wmma_bf16(qf1, load_frag(&Ks[cur][ar][32 + ko]),      s0);
    s1 = wmma_bf16(qf0, load_frag(&Ks[cur][16 + ar][ko]),      s1);
    s1 = wmma_bf16(qf1, load_frag(&Ks[cur][16 + ar][32 + ko]), s1);

    const float scale = 0.125f;          // 1/sqrt(64)
    int rofs = (lane & 16) ? 8 : 0;
#pragma unroll
    for (int j = 0; j < 8; ++j) {
      float a0 = s0[j] * scale, a1 = s1[j] * scale;
      float mx = fmaxf(a0, a1);
#pragma unroll
      for (int m = 8; m >= 1; m >>= 1) mx = fmaxf(mx, __shfl_xor(mx, m));
      float mn = fmaxf(mrow[j], mx);
      float f  = __expf(mrow[j] - mn);
      float p0 = __expf(a0 - mn), p1 = __expf(a1 - mn);
      float rs = p0 + p1;
#pragma unroll
      for (int m = 8; m >= 1; m >>= 1) rs += __shfl_xor(rs, m);
      lrow[j] = lrow[j] * f + rs;
      mrow[j] = mn;
      o0[j] *= f; o1[j] *= f; o2[j] *= f; o3[j] *= f;
      int row = j + rofs;
      Ps[w][row][ar]      = (bf16_t)p0;
      Ps[w][row][16 + ar] = (bf16_t)p1;
    }
    wait_ds0();                          // same-wave LDS RAW on Ps
    v16bf pf = load_frag(&Ps[w][ar][ko]);
    o0 = wmma_bf16(pf, load_frag(&Vt[cur][ar][ko]),      o0);
    o1 = wmma_bf16(pf, load_frag(&Vt[cur][16 + ar][ko]), o1);
    o2 = wmma_bf16(pf, load_frag(&Vt[cur][32 + ar][ko]), o2);
    o3 = wmma_bf16(pf, load_frag(&Vt[cur][48 + ar][ko]), o3);
    cur ^= 1;
  }

  // normalize + store to (B,T,D) bf16
  int b = bh / Hc, h = bh % Hc;
  int rofs = (lane & 16) ? 8 : 0;
#pragma unroll
  for (int j = 0; j < 8; ++j) {
    float inv = 1.f / lrow[j];
    int row = q0 + w * 16 + j + rofs;
    size_t rb = ((size_t)b * Tc + row) * Dc + h * Ec;
    obuf[rb + ar]      = (bf16_t)(o0[j] * inv);
    obuf[rb + 16 + ar] = (bf16_t)(o1[j] * inv);
    obuf[rb + 32 + ar] = (bf16_t)(o2[j] * inv);
    obuf[rb + 48 + ar] = (bf16_t)(o3[j] * inv);
  }
}

// ---------------------------------------------------------------------------
// Host-side launch.  Workspace layout (bytes), total ~184 MiB.
// ---------------------------------------------------------------------------
static const size_t OFF_MOD  = 0;                       //  8*3072 f32
static const size_t OFF_ST   = OFF_MOD  + 98304;        //  8*1024 f32
static const size_t OFF_WQKV = OFF_ST   + 32768;        //  1024*3072 bf16
static const size_t OFF_WOUT = OFF_WQKV + 6291456;      //  1024*1024 bf16
static const size_t OFF_H    = OFF_WOUT + 2097152;      //  8192*1024 bf16
static const size_t OFF_QKV  = OFF_H    + 16777216;     //  3*8*16*1024*64 f32
static const size_t OFF_QH   = OFF_QKV  + 100663296;    //  bf16
static const size_t OFF_KH   = OFF_QH   + 16777216;     //  bf16
static const size_t OFF_VH   = OFF_KH   + 16777216;     //  bf16
static const size_t OFF_O    = OFF_VH   + 16777216;     //  bf16

extern "C" void kernel_launch(void* const* d_in, const int* in_sizes, int n_in,
                              void* d_out, int out_size, void* d_ws, size_t ws_size,
                              hipStream_t stream) {
  (void)in_sizes; (void)n_in; (void)out_size; (void)ws_size;
  const float* x        = (const float*)d_in[0];
  const float* time_in  = (const float*)d_in[1];
  const float* position = (const float*)d_in[2];
  const float* mod_w    = (const float*)d_in[3];
  const float* mod_b    = (const float*)d_in[4];
  const float* w_qkv    = (const float*)d_in[5];
  const float* w_out    = (const float*)d_in[6];
  const float* qnw      = (const float*)d_in[7];
  const float* knw      = (const float*)d_in[8];
  float* out = (float*)d_out;

  char* ws = (char*)d_ws;
  float*  modb  = (float*)(ws + OFF_MOD);
  float*  st    = (float*)(ws + OFF_ST);
  bf16_t* wqkvb = (bf16_t*)(ws + OFF_WQKV);
  bf16_t* woutb = (bf16_t*)(ws + OFF_WOUT);
  bf16_t* hb    = (bf16_t*)(ws + OFF_H);
  float*  qkvb  = (float*)(ws + OFF_QKV);
  bf16_t* qh    = (bf16_t*)(ws + OFF_QH);
  bf16_t* kh    = (bf16_t*)(ws + OFF_KH);
  bf16_t* vh    = (bf16_t*)(ws + OFF_VH);
  bf16_t* ob    = (bf16_t*)(ws + OFF_O);

  silu_kernel<<<32, 256, 0, stream>>>(time_in, st, Bc * 1024);
  cvt_bf16_kernel<<<(Dc * N3c) / 256, 256, 0, stream>>>(w_qkv, wqkvb, Dc * N3c);
  cvt_bf16_kernel<<<(Dc * Dc) / 256, 256, 0, stream>>>(w_out, woutb, Dc * Dc);

  mod_kernel<<<dim3(N3c / 256, Bc), 256, 0, stream>>>(st, mod_w, mod_b, modb);

  ln_mod_kernel<<<Bc * Tc, 256, 0, stream>>>(x, modb, hb);

  gemm_bf16_kernel<<<dim3(N3c / 128, (Bc * Tc) / 128), 256, 0, stream>>>(
      hb, wqkvb, Bc * Tc, N3c, Dc, /*mode=*/0, qkvb, modb);

  qk_rope_v_kernel<<<(Bc * Hc * Tc) / 8, 256, 0, stream>>>(
      qkvb, position, qnw, knw, qh, kh, vh);

  attn_kernel<<<Bc * Hc * (Tc / 64), 128, 0, stream>>>(qh, kh, vh, ob);

  gemm_bf16_kernel<<<dim3(Dc / 128, (Bc * Tc) / 128), 256, 0, stream>>>(
      ob, woutb, Bc * Tc, Dc, Dc, /*mode=*/1, out, modb);
}
